// HeteroGAT_86955907875555
// MI455X (gfx1250) — matrix operand order repaired and checked by reading, not compiled, not run on hardware
//
#include <hip/hip_runtime.h>

// ---------------------------------------------------------------------------
// Types for CDNA5 WMMA (wave32)
// ---------------------------------------------------------------------------
typedef __attribute__((ext_vector_type(16))) __bf16 v16bf;
typedef __attribute__((ext_vector_type(8)))  float  v8f;

__device__ __forceinline__ __bf16 f2bf(float f) { return static_cast<__bf16>(f); }

// monotone float <-> ordered-u32 mapping (for atomicMax on floats)
__device__ __forceinline__ unsigned f2ord(float f) {
    union { float f; unsigned u; } a; a.f = f;
    unsigned u = a.u;
    return (u & 0x80000000u) ? ~u : (u | 0x80000000u);
}
__device__ __forceinline__ float ord2f(unsigned k) {
    unsigned u = (k & 0x80000000u) ? (k & 0x7FFFFFFFu) : ~k;
    union { unsigned u; float f; } a; a.u = u;
    return a.f;
}

// ---------------------------------------------------------------------------
// Zero-fill scratch / accumulators
// ---------------------------------------------------------------------------
__global__ void zero_u32(unsigned* __restrict__ p, unsigned n) {
    unsigned t = blockIdx.x * blockDim.x + threadIdx.x;
    if (t < n) p[t] = 0u;
}

// ---------------------------------------------------------------------------
// Y[M,N] = X[M,K] @ W[K,N]  (bf16 WMMA, f32 acc, one wave / 16x16 tile)
// K is a template parameter -> K-loop fully unrolled, branchless tails.
// Requires M % 16 == 0.  Columns n>=N load clamped addresses (never stored).
// ---------------------------------------------------------------------------
template <int K>
__global__ __launch_bounds__(32) void gemm_wmma_bf16(
        const float* __restrict__ X, const float* __restrict__ W,
        float* __restrict__ Y, int N) {
    const int lane = threadIdx.x & 31;
    const int l    = lane & 15;
    const int half = lane >> 4;                 // 0: lanes 0-15, 1: lanes 16-31
    const unsigned m0 = blockIdx.x * 16u;
    const unsigned n0 = blockIdx.y * 16u;
    const unsigned m  = m0 + (unsigned)l;       // always < M (M % 16 == 0)
    const int      n  = (int)n0 + l;
    const unsigned nc = (unsigned)((n < N) ? n : (N - 1));  // clamped column

    const float* __restrict__ Xr = X + (size_t)m * (unsigned)K;
    v8f acc = {};
    constexpr int KT = (K + 31) & ~31;
    #pragma unroll
    for (int k0 = 0; k0 < KT; k0 += 32) {
        const int kbA = k0 + half * 8;          // A: this lane's K-base
        const int kbB = k0 + half * 8;          // B: this lane's K-base
        v16bf a, b;
        if (k0 + 32 <= K) {
            // ---- full chunk: b128 vector loads, no masking ----
            const float4 a0 = *(const float4*)(Xr + kbA);
            const float4 a1 = *(const float4*)(Xr + kbA + 4);
            const float4 a2 = *(const float4*)(Xr + kbA + 16);
            const float4 a3 = *(const float4*)(Xr + kbA + 20);
            a[0]=f2bf(a0.x); a[1]=f2bf(a0.y); a[2]=f2bf(a0.z); a[3]=f2bf(a0.w);
            a[4]=f2bf(a1.x); a[5]=f2bf(a1.y); a[6]=f2bf(a1.z); a[7]=f2bf(a1.w);
            a[8]=f2bf(a2.x); a[9]=f2bf(a2.y); a[10]=f2bf(a2.z); a[11]=f2bf(a2.w);
            a[12]=f2bf(a3.x); a[13]=f2bf(a3.y); a[14]=f2bf(a3.z); a[15]=f2bf(a3.w);
            #pragma unroll
            for (int j = 0; j < 16; ++j)
                b[j] = f2bf(W[(unsigned)(kbB + j + ((j >= 8) ? 8 : 0)) * (unsigned)N + nc]);
        } else {
            // ---- tail chunk (compile-time known): clamped loads + selects ----
            #pragma unroll
            for (int j = 0; j < 8; ++j) {
                const int k1 = kbA + j, k2 = kbA + 16 + j;
                float v1 = Xr[(k1 < K) ? k1 : 0]; v1 = (k1 < K) ? v1 : 0.f;
                float v2 = Xr[(k2 < K) ? k2 : 0]; v2 = (k2 < K) ? v2 : 0.f;
                a[j]     = f2bf(v1);
                a[8 + j] = f2bf(v2);
            }
            #pragma unroll
            for (int j = 0; j < 16; ++j) {
                const int k = kbB + j + ((j >= 8) ? 8 : 0);
                float v = W[(unsigned)((k < K) ? k : 0) * (unsigned)N + nc];
                v = (k < K) ? v : 0.f;
                b[j] = f2bf(v);
            }
        }
        acc = __builtin_amdgcn_wmma_f32_16x16x32_bf16(
                  false, a, false, b, (short)0, acc, false, false);
    }
    if (n < N) {
        // C layout: VGPR j -> row m0+j (lanes 0-15) / m0+8+j (lanes 16-31)
        unsigned off = (m0 + (unsigned)(half * 8)) * (unsigned)N + (unsigned)n;
        #pragma unroll
        for (int j = 0; j < 8; ++j) {
            Y[off] = acc[j];
            off += (unsigned)N;
        }
    }
}

// ---------------------------------------------------------------------------
// qd[f,h] = sum_c Wd[f, h*ch+c] * ad[h,c];  kE[h] = sum_c We[h*ch+c]*ae[h,c]
// ---------------------------------------------------------------------------
__global__ void prep_qd(const float* __restrict__ Wd, const float* __restrict__ ad,
                        const float* __restrict__ We, const float* __restrict__ ae,
                        float* __restrict__ qd, float* __restrict__ kE,
                        int Fin, int H, int ch) {
    int t = blockIdx.x * blockDim.x + threadIdx.x;
    int HID = H * ch;
    if (t < Fin * H) {
        int f = t / H, h = t % H;
        float s = 0.f;
        for (int c = 0; c < ch; ++c)
            s += Wd[(unsigned)f * HID + h * ch + c] * ad[h * ch + c];
        qd[t] = s;
    }
    if (We != nullptr && t < H) {
        float s = 0.f;
        for (int c = 0; c < ch; ++c) s += We[t * ch + c] * ae[t * ch + c];
        kE[t] = s;
    }
}

// a_dst[n,h] = sum_f X[n,f] * qd[f,h]
__global__ void node_dot(const float* __restrict__ X, const float* __restrict__ qd,
                         float* __restrict__ out, int N, int Fin, int H) {
    int t = blockIdx.x * blockDim.x + threadIdx.x;
    if (t >= N * H) return;
    int n = t / H, h = t % H;
    const float* __restrict__ Xr = X + (size_t)n * (unsigned)Fin;
    float s = 0.f;
    for (int f = 0; f < Fin; ++f) s += Xr[f] * qd[f * H + h];
    out[t] = s;
}

// a_src[n,h] = sum_c hs[n, h*ch+c] * as[h,c]
__global__ void head_dot(const float* __restrict__ hs, const float* __restrict__ as_,
                         float* __restrict__ out, int N, int H, int ch) {
    int t = blockIdx.x * blockDim.x + threadIdx.x;
    if (t >= N * H) return;
    int n = t / H, h = t % H;
    const float* __restrict__ hr = hs + (size_t)n * (unsigned)(H * ch) + h * ch;
    float s = 0.f;
    for (int c = 0; c < ch; ++c) s += hr[c] * as_[h * ch + c];
    out[t] = s;
}

// logits + leaky relu + per-dst running max (ordered-u32 atomicMax)
__global__ void edge_logits(const int* __restrict__ src, const int* __restrict__ dst,
                            const float* __restrict__ ew, const float* __restrict__ kE,
                            const float* __restrict__ asrc, const float* __restrict__ adst,
                            float* __restrict__ elog, unsigned* __restrict__ mbuf,
                            int E, int H) {
    int t = blockIdx.x * blockDim.x + threadIdx.x;
    if (t >= E * H) return;
    int e = t / H, h = t % H;
    float lg = asrc[(unsigned)src[e] * H + h] + adst[(unsigned)dst[e] * H + h];
    if (ew != nullptr) lg += ew[e] * kE[h];
    lg = (lg > 0.f) ? lg : 0.2f * lg;          // leaky_relu(0.2)
    elog[t] = lg;
    atomicMax(&mbuf[(unsigned)dst[e] * H + h], f2ord(lg));
}

// e = exp(logit - max);  denom[dst] += e
__global__ void edge_expsum(const int* __restrict__ dst,
                            float* __restrict__ elog, const unsigned* __restrict__ mbuf,
                            float* __restrict__ den, int E, int H) {
    int t = blockIdx.x * blockDim.x + threadIdx.x;
    if (t >= E * H) return;
    int e = t / H, h = t % H;
    float m = ord2f(mbuf[(unsigned)dst[e] * H + h]);
    float v = __expf(elog[t] - m);
    elog[t] = v;
    atomicAdd(&den[(unsigned)dst[e] * H + h], v);
}

// out[dst, :] += hs[src, :] * alpha
// one wave per edge; lanes own float4-aligned channel chunks (HID % 4 == 0)
__global__ void edge_agg(const int* __restrict__ src, const int* __restrict__ dst,
                         const float* __restrict__ hs, const float* __restrict__ elog,
                         const float* __restrict__ den, float* __restrict__ out,
                         int E, int H, int ch) {
    unsigned g = blockIdx.x * blockDim.x + threadIdx.x;
    int e = (int)(g >> 5);
    if (e >= E) return;
    int lane = (int)(g & 31u);
    const unsigned s = (unsigned)src[e];
    const unsigned d = (unsigned)dst[e];
    const int HID = H * ch;
    const float* __restrict__ hr = hs + (size_t)s * (unsigned)HID;
    float* __restrict__ orow = out + (size_t)d * (unsigned)HID;
    const float* __restrict__ el = elog + (size_t)(unsigned)e * (unsigned)H;
    const float* __restrict__ dn = den + (size_t)d * (unsigned)H;
    for (int c0 = lane * 4; c0 < HID; c0 += 128) {
        const float4 v = *(const float4*)(hr + c0);
        #pragma unroll
        for (int j = 0; j < 4; ++j) {
            const int c = c0 + j;
            const int h = c / ch;
            const float alpha = el[h] / dn[h];
            const float x = (j == 0) ? v.x : (j == 1) ? v.y : (j == 2) ? v.z : v.w;
            atomicAdd(&orow[c], x * alpha);
        }
    }
}

// Y += bias; optional ELU
__global__ void bias_act(float* __restrict__ Y, const float* __restrict__ b,
                         unsigned total, int F, int elu) {
    unsigned t = blockIdx.x * blockDim.x + threadIdx.x;
    if (t >= total) return;
    int f = (int)(t % (unsigned)F);
    float v = Y[t] + b[f];
    if (elu) v = (v > 0.f) ? v : (__expf(v) - 1.f);
    Y[t] = v;
}

// ---------------------------------------------------------------------------
// Host-side stage driver (stream-ordered; graph-capture safe)
// ---------------------------------------------------------------------------
static inline int cdiv(long a, int b) { return (int)((a + b - 1) / b); }

static void launch_gemm(hipStream_t stream, const float* X, const float* W,
                        float* Y, int M, int N, int K) {
    dim3 gg((M + 15) / 16, (N + 15) / 16);
    if (K == 128)
        gemm_wmma_bf16<128><<<gg, 32, 0, stream>>>(X, W, Y, N);
    else  // K == 100 (layer-2 hidden width)
        gemm_wmma_bf16<100><<<gg, 32, 0, stream>>>(X, W, Y, N);
}

static void run_gat(hipStream_t stream,
                    const float* x_src, const float* x_dst,
                    const int* src, const int* dst, int E, const float* ew,
                    const float* Ws, const float* Wd,
                    const float* as_, const float* ad_,
                    const float* We, const float* ae, const float* bias,
                    int Ns, int Nd, int Fin, int H, int ch,
                    float* hs, float* asrc, float* adst,
                    unsigned* mbuf, float* den, float* elog,
                    float* qd, float* kE,
                    float* out, int elu) {
    const int HID = H * ch;
    // 0) tiny precomputes: qd [Fin,H], kE [H]
    prep_qd<<<cdiv((long)Fin * H, 256), 256, 0, stream>>>(Wd, ad_, We, ae, qd, kE, Fin, H, ch);
    // 1) hs = x_src @ Ws  (WMMA bf16)
    launch_gemm(stream, x_src, Ws, hs, Ns, HID, Fin);
    // 2) attention scalars
    head_dot<<<cdiv((long)Ns * H, 256), 256, 0, stream>>>(hs, as_, asrc, Ns, H, ch);
    node_dot<<<cdiv((long)Nd * H, 256), 256, 0, stream>>>(x_dst, qd, adst, Nd, Fin, H);
    // 3) zero max / denom / output accumulator
    zero_u32<<<cdiv((long)Nd * H, 256), 256, 0, stream>>>(mbuf, (unsigned)(Nd * H));
    zero_u32<<<cdiv((long)Nd * H, 256), 256, 0, stream>>>((unsigned*)den, (unsigned)(Nd * H));
    zero_u32<<<cdiv((long)Nd * HID, 256), 256, 0, stream>>>((unsigned*)out, (unsigned)(Nd * HID));
    // 4) edge pipeline: logits+max -> exp+sum -> aggregate
    edge_logits<<<cdiv((long)E * H, 256), 256, 0, stream>>>(src, dst, ew, kE, asrc, adst, elog, mbuf, E, H);
    edge_expsum<<<cdiv((long)E * H, 256), 256, 0, stream>>>(dst, elog, mbuf, den, E, H);
    edge_agg<<<cdiv((long)E * 32, 256), 256, 0, stream>>>(src, dst, hs, elog, den, out, E, H, ch);
    // 5) bias (+ELU for layer 1)
    bias_act<<<cdiv((long)Nd * HID, 256), 256, 0, stream>>>(out, bias, (unsigned)(Nd * HID), HID, elu);
}

extern "C" void kernel_launch(void* const* d_in, const int* in_sizes, int n_in,
                              void* d_out, int out_size, void* d_ws, size_t ws_size,
                              hipStream_t stream) {
    const float* x_A  = (const float*)d_in[0];
    const float* x_B  = (const float*)d_in[1];
    const int*   eiAB = (const int*)d_in[2];
    const int*   eiBA = (const int*)d_in[3];
    const float* w_AB = (const float*)d_in[4];
    const float* w_BA = (const float*)d_in[5];

    const int E   = in_sizes[4];
    const int HID = in_sizes[12];              // l1_AB_b -> 100
    const int F   = in_sizes[6] / HID;         // l1_AB_Ws (F,HID) -> 128
    const int NA  = in_sizes[0] / F;
    const int NB  = in_sizes[1] / F;
    const int H1 = 4, ch1 = HID / 4;
    const int H2 = 1, ch2 = HID;
    const int Nmax = (NA > NB) ? NA : NB;

    // parameter pointers (setup_inputs order: per et -> l1 block then l2 block)
    #define P(i) ((const float*)d_in[(i)])
    const float *l1AB_Ws=P(6),  *l1AB_Wd=P(7),  *l1AB_as=P(8),  *l1AB_ad=P(9),
                *l1AB_We=P(10), *l1AB_ae=P(11), *l1AB_b=P(12);
    const float *l2AB_Ws=P(13), *l2AB_Wd=P(14), *l2AB_as=P(15), *l2AB_ad=P(16), *l2AB_b=P(17);
    const float *l1BA_Ws=P(18), *l1BA_Wd=P(19), *l1BA_as=P(20), *l1BA_ad=P(21),
                *l1BA_We=P(22), *l1BA_ae=P(23), *l1BA_b=P(24);
    const float *l2BA_Ws=P(25), *l2BA_Wd=P(26), *l2BA_as=P(27), *l2BA_ad=P(28), *l2BA_b=P(29);
    #undef P

    // workspace layout (floats)
    float* w = (float*)d_ws;
    const size_t nHS = (size_t)Nmax * HID;     // 5,000,000
    const size_t nNH = (size_t)Nmax * H1;      // 200,000 (max H)
    float*    hs   = w;             w += nHS;
    float*    hA   = w;             w += nHS;  // ELU'd layer-1 output on A side
    float*    hB   = w;             w += nHS;  // ELU'd layer-1 output on B side
    float*    asrc = w;             w += nNH;
    float*    adst = w;             w += nNH;
    unsigned* mbuf = (unsigned*)w;  w += nNH;
    float*    den  = w;             w += nNH;
    float*    elog = w;             w += (size_t)E * H1;
    float*    qd   = w;             w += 1024;
    float*    kE   = w;             w += 16;

    const int* sAB = eiAB;           const int* dAB = eiAB + E;
    const int* sBA = eiBA;           const int* dBA = eiBA + E;

    float* oA = (float*)d_out;                         // [NA, HID]
    float* oB = (float*)d_out + (size_t)NA * HID;      // [NB, HID]

    // ---- Layer 1 (heads=4, ch=25, edge-weighted, ELU) ----
    run_gat(stream, x_A, x_B, sAB, dAB, E, w_AB,
            l1AB_Ws, l1AB_Wd, l1AB_as, l1AB_ad, l1AB_We, l1AB_ae, l1AB_b,
            NA, NB, F, H1, ch1, hs, asrc, adst, mbuf, den, elog, qd, kE, hB, 1);
    run_gat(stream, x_B, x_A, sBA, dBA, E, w_BA,
            l1BA_Ws, l1BA_Wd, l1BA_as, l1BA_ad, l1BA_We, l1BA_ae, l1BA_b,
            NB, NA, F, H1, ch1, hs, asrc, adst, mbuf, den, elog, qd, kE, hA, 1);
    // ---- Layer 2 (heads=1, ch=100, no edge features, bias only) ----
    run_gat(stream, hA, hB, sAB, dAB, E, nullptr,
            l2AB_Ws, l2AB_Wd, l2AB_as, l2AB_ad, nullptr, nullptr, l2AB_b,
            NA, NB, HID, H2, ch2, hs, asrc, adst, mbuf, den, elog, qd, kE, oB, 0);
    run_gat(stream, hB, hA, sBA, dBA, E, nullptr,
            l2BA_Ws, l2BA_Wd, l2BA_as, l2BA_ad, nullptr, nullptr, l2BA_b,
            NB, NA, HID, H2, ch2, hs, asrc, adst, mbuf, den, elog, qd, kE, oA, 0);
}
